// GATLayer_45853070852633
// MI455X (gfx1250) — compile-verified, matrix operand order checked by reading, not run on hardware
//
#include <hip/hip_runtime.h>
#include <hip/hip_bf16.h>

typedef __attribute__((ext_vector_type(16))) _Float16 v16h;
typedef __attribute__((ext_vector_type(8)))  float    v8f;

#define IN_F      64
#define OUT_F     32
#define NEG_SLOPE 0.2f

// ---- monotone float <-> uint encoding for atomicMax on floats ----
__device__ __forceinline__ unsigned enc_f32(float f) {
  unsigned u = __float_as_uint(f);
  return (u & 0x80000000u) ? ~u : (u | 0x80000000u);
}
__device__ __forceinline__ float dec_f32(unsigned u) {
  return (u & 0x80000000u) ? __uint_as_float(u & 0x7FFFFFFFu)
                           : __uint_as_float(~u);
}

// -------------------------------------------------------------------------
// Kernel 0: pre-pack W[64,32] f32 -> f16 in the exact WMMA B-register
// layout: fragment (t, ks) for lane L lives at wp[((t*2+ks)*32 + L)*16 .. +15]
// (32 contiguous bytes -> two b128 loads in the GEMM).
// B layout (16-bit 32x16 KxN): lane = column, halves j map to
//   K = kbase + (j<8 ? j : 8+j), kbase = (lane>=16 ? 8 : 0).
// -------------------------------------------------------------------------
__global__ __launch_bounds__(128)
void k_pack_w(const float* __restrict__ W, _Float16* __restrict__ wp) {
  const int tid   = threadIdx.x;          // single 128-thread block
  const int lane  = tid & 31;
  const int ks    = (tid >> 5) & 1;
  const int t     = tid >> 6;
  const int kbase = (lane >= 16) ? 8 : 0;
  const int n     = t * 16 + (lane & 15);
  _Float16* dst = wp + (size_t)((t * 2 + ks) * 32 + lane) * 16;
#pragma unroll
  for (int j = 0; j < 8; ++j) {
    dst[j]     = (_Float16)W[(size_t)(ks * 32 + kbase + j)      * OUT_F + n];
    dst[j + 8] = (_Float16)W[(size_t)(ks * 32 + kbase + 16 + j) * OUT_F + n];
  }
}

// -------------------------------------------------------------------------
// Kernel 1: h[N,32] = x[N,64] @ W[64,32] via v_wmma_f32_16x16x32_f16.
// One wave per 16-row tile; 2 K-steps x 2 N-tiles = 4 WMMAs.
// A loads vectorized as float4 (b128); B fragments are pre-packed f16.
// -------------------------------------------------------------------------
__global__ __launch_bounds__(128)
void k_gemm(const float* __restrict__ x, const _Float16* __restrict__ wpack,
            float* __restrict__ h, int nNodes) {
  const int lane = threadIdx.x & 31;
  const int wave = blockIdx.x * (blockDim.x >> 5) + (threadIdx.x >> 5);
  const int m0   = wave * 16;
  if (m0 >= nNodes) return;              // wave-uniform exit: EXEC all-1 at WMMA

  const int mrow  = lane & 15;
  const int kbase = (lane >= 16) ? 8 : 0;
  int row = m0 + mrow;
  if (row > nNodes - 1) row = nNodes - 1;

  // ---- A: two K-steps of 32, vectorized loads + f16 convert ----
  v16h a[2];
#pragma unroll
  for (int ks = 0; ks < 2; ++ks) {
    const float4* xr = (const float4*)(x + (size_t)row * IN_F + ks * 32 + kbase);
    const float4 lo0 = xr[0], lo1 = xr[1];   // K = kbase + 0..7
    const float4 hi0 = xr[4], hi1 = xr[5];   // K = kbase + 16..23
    a[ks][0]  = (_Float16)lo0.x; a[ks][1]  = (_Float16)lo0.y;
    a[ks][2]  = (_Float16)lo0.z; a[ks][3]  = (_Float16)lo0.w;
    a[ks][4]  = (_Float16)lo1.x; a[ks][5]  = (_Float16)lo1.y;
    a[ks][6]  = (_Float16)lo1.z; a[ks][7]  = (_Float16)lo1.w;
    a[ks][8]  = (_Float16)hi0.x; a[ks][9]  = (_Float16)hi0.y;
    a[ks][10] = (_Float16)hi0.z; a[ks][11] = (_Float16)hi0.w;
    a[ks][12] = (_Float16)hi1.x; a[ks][13] = (_Float16)hi1.y;
    a[ks][14] = (_Float16)hi1.z; a[ks][15] = (_Float16)hi1.w;
  }

  // ---- B: pre-packed fragments, 32B contiguous per lane ----
  const v16h* wp = (const v16h*)wpack;
  v8f acc[2] = {};
#pragma unroll
  for (int t = 0; t < 2; ++t)
#pragma unroll
    for (int ks = 0; ks < 2; ++ks) {
      const v16h b = wp[(t * 2 + ks) * 32 + lane];
      acc[t] = __builtin_amdgcn_wmma_f32_16x16x32_f16(
          false, a[ks], false, b, (short)0, acc[t], false, false);
    }

  // ---- store D: per VGPR r, 16 lanes cover 16 contiguous columns ----
  const int mi = m0 + ((lane >= 16) ? 8 : 0);
#pragma unroll
  for (int t = 0; t < 2; ++t)
#pragma unroll
    for (int r = 0; r < 8; ++r)
      h[(size_t)(mi + r) * OUT_F + t * 16 + mrow] = acc[t][r];
}

// -------------------------------------------------------------------------
// Kernel 2: per-node attention logits + zero-init of reduction buffers.
// -------------------------------------------------------------------------
__global__ __launch_bounds__(256)
void k_node_prep(const float* __restrict__ h,
                 const float* __restrict__ att_src,
                 const float* __restrict__ att_dst,
                 float* __restrict__ a_src, float* __restrict__ a_dst,
                 unsigned* __restrict__ m_enc, float* __restrict__ denom,
                 float* __restrict__ out, int nNodes) {
  const int i = blockIdx.x * blockDim.x + threadIdx.x;
  if (i >= nNodes) return;
  const float4* hv = (const float4*)(h + (size_t)i * OUT_F);
  float s = 0.f, d = 0.f;
#pragma unroll
  for (int q = 0; q < 8; ++q) {
    const float4 hh = hv[q];
    s += hh.x * att_src[q * 4 + 0] + hh.y * att_src[q * 4 + 1] +
         hh.z * att_src[q * 4 + 2] + hh.w * att_src[q * 4 + 3];
    d += hh.x * att_dst[q * 4 + 0] + hh.y * att_dst[q * 4 + 1] +
         hh.z * att_dst[q * 4 + 2] + hh.w * att_dst[q * 4 + 3];
  }
  a_src[i] = s;
  a_dst[i] = d;
  m_enc[i] = 0u;          // below enc of any finite float; self-loop updates it
  denom[i] = 0.f;
  const float4 z = {0.f, 0.f, 0.f, 0.f};
  float4* ov = (float4*)(out + (size_t)i * OUT_F);
#pragma unroll
  for (int q = 0; q < 8; ++q) ov[q] = z;
}

// -------------------------------------------------------------------------
// Kernel 3: edge pass 1 — leaky-relu logit, segment max via atomicMax(u32).
// Edges [0,E) from edge_index; edges [E,E+N) are self loops.
// -------------------------------------------------------------------------
__global__ __launch_bounds__(256)
void k_edge_max(const int* __restrict__ ei, long long E, int nNodes,
                const float* __restrict__ a_src, const float* __restrict__ a_dst,
                float* __restrict__ ebuf, unsigned* __restrict__ m_enc) {
  const long long e = (long long)blockIdx.x * blockDim.x + threadIdx.x;
  const long long tot = E + nNodes;
  if (e >= tot) return;
  int s, d;
  if (e < E) { s = ei[e]; d = ei[E + e]; } else { s = d = (int)(e - E); }
  float v = a_src[s] + a_dst[d];
  v = (v > 0.f) ? v : NEG_SLOPE * v;
  ebuf[e] = v;
  atomicMax(m_enc + d, enc_f32(v));
}

// -------------------------------------------------------------------------
// Kernel 4: edge pass 2 — wave per edge, lane per channel.
// ex = exp(e - m[dst]) (wave-uniform); lane 0 accumulates the softmax
// denominator; all 32 lanes scatter unnormalized ex * h[src][lane] into out.
// Normalization (/denom) happens once per node in the finalize kernel —
// algebraically identical to per-edge alpha = ex/denom.
// -------------------------------------------------------------------------
__global__ __launch_bounds__(256)
void k_edge_scatter(const int* __restrict__ ei, long long E, int nNodes,
                    const float* __restrict__ ebuf,
                    const unsigned* __restrict__ m_enc,
                    float* __restrict__ denom,
                    const float* __restrict__ h, float* __restrict__ out) {
  const int lane = threadIdx.x & 31;
  const int wv   = __builtin_amdgcn_readfirstlane(threadIdx.x >> 5);
  const long long e = (long long)blockIdx.x * (blockDim.x >> 5) + wv;
  const long long tot = E + nNodes;
  if (e >= tot) return;
  int s, d;
  if (e < E) { s = ei[e]; d = ei[E + e]; } else { s = d = (int)(e - E); }
  const float ex = __expf(ebuf[e] - dec_f32(m_enc[d]));
  if (lane == 0) atomicAdd(denom + d, ex);
  const float val = h[(size_t)s * OUT_F + lane] * ex;
  atomicAdd(out + (size_t)d * OUT_F + lane, val);
}

// -------------------------------------------------------------------------
// Kernel 5: normalize by softmax denominator, add bias, ReLU.
// -------------------------------------------------------------------------
__global__ __launch_bounds__(256)
void k_finalize(float* __restrict__ out, const float* __restrict__ denom,
                const float* __restrict__ bias, long long total) {
  const long long idx = (long long)blockIdx.x * blockDim.x + threadIdx.x;
  if (idx >= total) return;
  const int node = (int)(idx >> 5);
  const int c    = (int)(idx & (OUT_F - 1));
  const float v = out[idx] / denom[node] + bias[c];
  out[idx] = v > 0.f ? v : 0.f;
}

// -------------------------------------------------------------------------
extern "C" void kernel_launch(void* const* d_in, const int* in_sizes, int n_in,
                              void* d_out, int out_size, void* d_ws, size_t ws_size,
                              hipStream_t stream) {
  const float* x       = (const float*)d_in[0];
  const int*   ei      = (const int*)  d_in[1];   // [2, E] flat
  const float* W       = (const float*)d_in[2];
  const float* att_src = (const float*)d_in[3];
  const float* att_dst = (const float*)d_in[4];
  const float* bias    = (const float*)d_in[5];
  float*       out     = (float*)d_out;

  const int       nNodes = in_sizes[0] / IN_F;
  const long long E      = (long long)in_sizes[1] / 2;
  const long long tot    = E + nNodes;

  // ---- workspace carve-up (each segment 256B-aligned) ----
  char* ws = (char*)d_ws;
  auto carve = [&](size_t bytes) {
    char* p = ws;
    ws += (bytes + 255) & ~(size_t)255;
    return p;
  };
  float*     h     = (float*)    carve((size_t)nNodes * OUT_F * sizeof(float));
  float*     a_src = (float*)    carve((size_t)nNodes * sizeof(float));
  float*     a_dst = (float*)    carve((size_t)nNodes * sizeof(float));
  unsigned*  m_enc = (unsigned*) carve((size_t)nNodes * sizeof(unsigned));
  float*     denom = (float*)    carve((size_t)nNodes * sizeof(float));
  _Float16*  wpack = (_Float16*) carve((size_t)4 * 32 * 16 * sizeof(_Float16));
  float*     ebuf  = (float*)    carve((size_t)tot * sizeof(float));

  // 0) pack W into WMMA B-fragment layout (f16)
  k_pack_w<<<1, 128, 0, stream>>>(W, wpack);

  // 1) projection (WMMA)
  const int wavesGemm  = (nNodes + 15) / 16;
  const int blocksGemm = (wavesGemm + 3) / 4;      // 4 waves / 128-thread block
  k_gemm<<<blocksGemm, 128, 0, stream>>>(x, wpack, h, nNodes);

  // 2) node logits + init
  k_node_prep<<<(nNodes + 255) / 256, 256, 0, stream>>>(
      h, att_src, att_dst, a_src, a_dst, m_enc, denom, out, nNodes);

  // 3) segment max
  k_edge_max<<<(int)((tot + 255) / 256), 256, 0, stream>>>(
      ei, E, nNodes, a_src, a_dst, ebuf, m_enc);

  // 4) exp + denom accumulation + weighted scatter (wave per edge)
  k_edge_scatter<<<(int)((tot + 7) / 8), 256, 0, stream>>>(
      ei, E, nNodes, ebuf, m_enc, denom, h, out);

  // 5) normalize + bias + relu
  const long long totalOut = (long long)nNodes * OUT_F;
  k_finalize<<<(int)((totalOut + 255) / 256), 256, 0, stream>>>(
      out, denom, bias, totalOut);
}
